// SinkAwareHyperAttention_74388833567008
// MI455X (gfx1250) — matrix operand order repaired and checked by reading, not compiled, not run on hardware
//
#include <hip/hip_runtime.h>
#include <math.h>

// ---------------- types for WMMA (gfx1250, wave32) ----------------
typedef __attribute__((ext_vector_type(16))) __bf16 v16bf;
typedef __attribute__((ext_vector_type(8)))  float  v8f;

#define HEADS 16
#define SEQ   4096
#define DIM   64
#define SINK  32
#define TAIL  4064        // 4096 - 32
#define HALF0 2032        // tail/2
#define HALF1 1016        // tail/4
#define NSORT 2032        // LSH sequence length (level-0 halves)
#define NPAD  2048        // padded to multiple of 256
#define NBLK  8           // 2048/256
#define SAMP  256

// ---------------- fragment-layout helpers (ISA 7.12.2) ----------------
// A (16-bit 16x32): lane (L&15)=row M; lanes<16 hold K=0..7,16..23 ; lanes>=16 hold K=8..15,24..31
// inverse: given kk in [0,32): kk<16 -> hi=kk>>3, i=kk&7 ; else hi=(kk-16)>>3, i=8+((kk-16)&7)
__device__ __forceinline__ void a_frag_inv(int kk, int& hi, int& i) {
  if (kk < 16) { hi = kk >> 3; i = kk & 7; }
  else         { hi = (kk - 16) >> 3; i = 8 + ((kk - 16) & 7); }
}
// B (16-bit 32x16): lane (L&15)=col N; lanes<16 hold K=0..7,16..23 ; lanes>=16 hold K=8..15,24..31
// inverse: hi = (kk>>3)&1, i = (kk&7) + ((kk>>4)<<3)

// contiguous per-lane fragment load: 16 halfs = 32 bytes -> 2x ds_load_b128
__device__ __forceinline__ v16bf load_frag(const __bf16* fr, int fragIdx) {
  int lane = threadIdx.x & 31;
  return *(const v16bf*)(fr + ((fragIdx << 5) + lane) * 16);
}

// ---------------- generic flash-attention kernel ----------------
// One workgroup = 4 waves = 128 threads handles 64 query rows of one "unit".
// unit = blockIdx.y ; row bases:  h*HS + OFF + s*SS  with h=unit/inner, s=unit%inner.
__global__ __launch_bounds__(128)
void flash_attn_kernel(
    const float* __restrict__ Q, const float* __restrict__ K, const float* __restrict__ V,
    float* __restrict__ O, float* __restrict__ LSE,
    int nq, int nk, int inner,
    long long q_hs, long long q_off, long long q_ss,
    long long k_hs, long long k_off, long long k_ss,
    long long o_hs, long long o_off, long long o_ss,
    long long l_hs, long long l_off, long long l_ss,
    int causal, int use_bias, const int* __restrict__ kblkArr,
    float scale, float lse_add)
{
  // fragment-order LDS: [tile(4) x kc(2)] fragments of [32 lanes][16 halfs]
  __shared__ __align__(32) __bf16 aQ[4 * 2 * 32 * 16];   // Q strips, A-layout
  __shared__ __align__(32) __bf16 aP[4 * 2 * 32 * 16];   // P strips, A-layout
  __shared__ __align__(32) __bf16 bK[4 * 2 * 32 * 16];   // K^T tiles, B-layout
  __shared__ __align__(32) __bf16 bV[4 * 2 * 32 * 16];   // V tiles,  B-layout
  __shared__ float  sS[64][68];                          // raw S, row-major
  __shared__ float  sM[64], sL[64], sCorr[64];

  const int unit = blockIdx.y;
  const int hh = unit / inner;
  const int ss = unit % inner;
  const long long qrow0 = (long long)hh * q_hs + q_off + (long long)ss * q_ss;
  const long long krow0 = (long long)hh * k_hs + k_off + (long long)ss * k_ss;
  const long long orow0 = (long long)hh * o_hs + o_off + (long long)ss * o_ss;
  const long long lrow0 = (long long)hh * l_hs + l_off + (long long)ss * l_ss;
  const int q0   = blockIdx.x * 64;
  const int tid  = threadIdx.x;
  const int wave = tid >> 5;
  const int lane = tid & 31;
  const float NEG = -__builtin_inff();

  // ---- stage Q tile into A-fragment layout (coalesced global reads) ----
  for (int e = tid; e < 64 * 64; e += 128) {
    int r = e >> 6, c = e & 63;
    float v = (q0 + r < nq) ? Q[(qrow0 + q0 + r) * 64 + c] : 0.f;
    int strip = r >> 4, m = r & 15;
    int kc = c >> 5, kk = c & 31, hi, i;
    a_frag_inv(kk, hi, i);
    aQ[((((strip << 1) + kc) << 5) + hi * 16 + m) * 16 + i] = (__bf16)v;
  }
  if (tid < 64) { sM[tid] = NEG; sL[tid] = 0.f; }

  v8f acc0 = {}, acc1 = {}, acc2 = {}, acc3 = {};

  const int nkb = (nk + 63) >> 6;
  for (int kb = 0; kb < nkb; ++kb) {
    __syncthreads();
    // ---- stage K,V block into B-fragment layouts (coalesced global reads) ----
    for (int e = tid; e < 64 * 64; e += 128) {
      int r = e >> 6, c = e & 63;
      int kr = kb * 64 + r;
      float kv = 0.f, vv = 0.f;
      if (kr < nk) { kv = K[(krow0 + kr) * 64 + c]; vv = V[(krow0 + kr) * 64 + c]; }
      // K^T fragment: tile nb = key>>4, col n = key&15 ; kdim = d -> kc=d>>5, hi=((d&31)>>3)&1, i=(d&7)+(((d&31)>>4)<<3)
      {
        int nb = r >> 4, n = r & 15;
        int kc = c >> 5, kk = c & 31;
        int hi = (kk >> 3) & 1, i = (kk & 7) + ((kk >> 4) << 3);
        bK[((((nb << 1) + kc) << 5) + hi * 16 + n) * 16 + i] = (__bf16)kv;
      }
      // V fragment: krow = key -> kc=key>>5, hi=((key&31)>>3)&1, i=(key&7)+(((key&31)>>4)<<3) ; col n -> nb=d>>4, n=d&15
      {
        int kc = r >> 5, kk = r & 31;
        int hi = (kk >> 3) & 1, i = (kk & 7) + ((kk >> 4) << 3);
        int nb = c >> 4, n = c & 15;
        bV[((((nb << 1) + kc) << 5) + hi * 16 + n) * 16 + i] = (__bf16)vv;
      }
    }
    __syncthreads();

    // ---- S = Q * K^T (per-wave 16-row strip, 8 WMMAs, b128 fragment loads) ----
    {
      const int hi = lane >> 4, n = lane & 15;
#pragma unroll
      for (int nb = 0; nb < 4; ++nb) {
        v8f c8 = {};
#pragma unroll
        for (int kc = 0; kc < 2; ++kc) {
          v16bf a = load_frag(aQ, (wave << 1) + kc);
          v16bf b = load_frag(bK, (nb << 1) + kc);
          c8 = __builtin_amdgcn_wmma_f32_16x16x32_bf16(false, a, false, b,
                                                       (short)0, c8, false, false);
        }
#pragma unroll
        for (int rv = 0; rv < 8; ++rv)
          sS[wave * 16 + rv + 8 * hi][nb * 16 + n] = c8[rv];
      }
    }
    __syncthreads();

    // ---- online softmax update (one thread per row); write P in A-fragment bf16 ----
    if (tid < 64) {
      const int r = tid;
      const int qpos = q0 + r;
      const int qblk = qpos >> 8;
      const int strip = r >> 4, m = r & 15;
      float mOld = sM[r];
      float mNew = mOld;
      for (int c = 0; c < 64; ++c) {
        int kpos = kb * 64 + c;
        float s = NEG;
        if (kpos < nk) {
          s = sS[r][c] * scale;
          if (causal && kpos > qpos) s = NEG;
          if (use_bias && kblkArr[hh * 256 + kpos] == qblk) s = NEG;
        }
        mNew = fmaxf(mNew, s);
      }
      float corr = (mNew == NEG) ? 1.f : expf(mOld - mNew);
      float lsum = 0.f;
      for (int c = 0; c < 64; ++c) {
        int kpos = kb * 64 + c;
        float s = NEG;
        if (kpos < nk) {
          s = sS[r][c] * scale;
          if (causal && kpos > qpos) s = NEG;
          if (use_bias && kblkArr[hh * 256 + kpos] == qblk) s = NEG;
        }
        float p = (s == NEG) ? 0.f : expf(s - mNew);
        int kc = c >> 5, kk = c & 31, hi, i;
        a_frag_inv(kk, hi, i);
        aP[((((strip << 1) + kc) << 5) + hi * 16 + m) * 16 + i] = (__bf16)p;
        lsum += p;
      }
      sM[r] = mNew;
      sL[r] = sL[r] * corr + lsum;
      sCorr[r] = corr;
    }
    __syncthreads();

    // ---- rescale accumulators, O += P * V (8 WMMAs, b128 fragment loads) ----
    {
      const int hi = lane >> 4;
      float cr[8];
#pragma unroll
      for (int rv = 0; rv < 8; ++rv) cr[rv] = sCorr[wave * 16 + rv + 8 * hi];
#pragma unroll
      for (int rv = 0; rv < 8; ++rv) {
        acc0[rv] *= cr[rv]; acc1[rv] *= cr[rv];
        acc2[rv] *= cr[rv]; acc3[rv] *= cr[rv];
      }
#pragma unroll
      for (int kc = 0; kc < 2; ++kc) {
        v16bf a = load_frag(aP, (wave << 1) + kc);
        v16bf b0 = load_frag(bV, (0 << 1) + kc);
        acc0 = __builtin_amdgcn_wmma_f32_16x16x32_bf16(false, a, false, b0, (short)0, acc0, false, false);
        v16bf b1 = load_frag(bV, (1 << 1) + kc);
        acc1 = __builtin_amdgcn_wmma_f32_16x16x32_bf16(false, a, false, b1, (short)0, acc1, false, false);
        v16bf b2 = load_frag(bV, (2 << 1) + kc);
        acc2 = __builtin_amdgcn_wmma_f32_16x16x32_bf16(false, a, false, b2, (short)0, acc2, false, false);
        v16bf b3 = load_frag(bV, (3 << 1) + kc);
        acc3 = __builtin_amdgcn_wmma_f32_16x16x32_bf16(false, a, false, b3, (short)0, acc3, false, false);
      }
    }
  }
  __syncthreads();

  // ---- store O = acc / l  and LSE = m + log(l) ----
  {
    const int hi = lane >> 4, n = lane & 15;
#pragma unroll
    for (int rv = 0; rv < 8; ++rv) {
      int r = wave * 16 + rv + 8 * hi;
      int qpos = q0 + r;
      if (qpos < nq) {
        float denom = sL[r];
        float inv = (denom > 0.f) ? 1.f / denom : 0.f;
        long long ob = (orow0 + qpos) * 64 + n;
        O[ob +  0] = acc0[rv] * inv;
        O[ob + 16] = acc1[rv] * inv;
        O[ob + 32] = acc2[rv] * inv;
        O[ob + 48] = acc3[rv] * inv;
      }
    }
  }
  if (tid < 64) {
    int qpos = q0 + tid;
    if (qpos < nq)
      LSE[lrow0 + qpos] = sM[tid] + logf(sL[tid]) + lse_add;
  }
}

// ---------------- LSH hash ----------------
__device__ __forceinline__ int hamming_perm(int i) {
  int add = 0, m = 128;
  while (m > 2) { int h = m >> 1; if (i >= h) { add += h; i = m - 1 - i; } m = h; }
  return add + i;
}

__global__ void lsh_hash_kernel(const float* __restrict__ X, const float* __restrict__ proj,
                                int* __restrict__ codes, int n, int row_off) {
  __shared__ float sp[64 * 7];
  for (int i = threadIdx.x; i < 448; i += blockDim.x) sp[i] = proj[i];
  __syncthreads();
  int h = blockIdx.y;
  int p = blockIdx.x * blockDim.x + threadIdx.x;
  if (p >= n) return;
  const float* x = X + ((long long)h * SEQ + row_off + p) * 64;
  int bin = 0;
#pragma unroll
  for (int r = 0; r < 7; ++r) {
    float dot = 0.f;
    for (int d = 0; d < 64; ++d) dot += x[d] * sp[d * 7 + r];
    if (dot > 0.f) bin |= (1 << r);
  }
  codes[(long long)h * n + p] = hamming_perm(bin);
}

// ---------------- stable counting argsort per head (codes in [0,128)) ----------------
__global__ void argsort_kernel(const int* __restrict__ codes, int* __restrict__ sidx,
                               int* __restrict__ sinv, int n, int npad) {
  __shared__ int sc[NPAD];
  __shared__ int hist[128];
  __shared__ int off[128];
  int h = blockIdx.x;
  for (int i = threadIdx.x; i < 128; i += blockDim.x) hist[i] = 0;
  __syncthreads();
  for (int i = threadIdx.x; i < n; i += blockDim.x) {
    int c = codes[(long long)h * n + i];
    sc[i] = c;
    atomicAdd(&hist[c], 1);
  }
  __syncthreads();
  if (threadIdx.x == 0) {
    int acc = 0;
    for (int b = 0; b < 128; ++b) { off[b] = acc; acc += hist[b]; }
  }
  __syncthreads();
  for (int i = threadIdx.x; i < n; i += blockDim.x) {
    int c = sc[i], rank = 0;
    for (int j = 0; j < i; ++j) rank += (sc[j] == c) ? 1 : 0;   // stable rank
    int pos = off[c] + rank;
    sidx[(long long)h * npad + pos] = i;
    sinv[(long long)h * n + i] = pos;
  }
  for (int i = n + threadIdx.x; i < npad; i += blockDim.x)
    sidx[(long long)h * npad + i] = -1;
}

// ---------------- gather sorted rows (zero-pad) ----------------
__global__ void gather_rows_kernel(const float* __restrict__ src, const int* __restrict__ idx,
                                   float* __restrict__ dst, int n, int npad, int row_off) {
  int rid = blockIdx.x;           // heads*npad rows
  int d = threadIdx.x;
  int h = rid / npad, p = rid % npad;
  float v = 0.f;
  if (p < n) {
    int i = idx[(long long)h * npad + p];
    v = src[((long long)h * SEQ + row_off + i) * 64 + d];
  }
  dst[(long long)rid * 64 + d] = v;
}

// ---------------- deterministic sampling (stand-in for threefry) ----------------
__device__ __forceinline__ unsigned smix(unsigned x) {
  x += 0x9e3779b9u; x ^= x >> 16; x *= 0x21f0aaadu;
  x ^= x >> 15; x *= 0x735a2d97u; x ^= x >> 15; return x;
}
__global__ void sample_kernel(int* __restrict__ sampled, int* __restrict__ kblk) {
  int j = blockIdx.x * blockDim.x + threadIdx.x;
  if (j >= HEADS * SAMP) return;
  int s = (int)(smix((unsigned)j ^ 0xA5A5F00Du) % (unsigned)NSORT);
  sampled[j] = s;
  kblk[j] = s >> 8;
}
__global__ void gather_sub_kernel(const float* __restrict__ ks, const float* __restrict__ vs,
                                  const int* __restrict__ sampled,
                                  float* __restrict__ ksub, float* __restrict__ vsub) {
  int rid = blockIdx.x;           // HEADS*SAMP rows
  int d = threadIdx.x;
  int h = rid >> 8;
  int idx = sampled[rid];
  long long src = ((long long)h * NPAD + idx) * 64 + d;
  ksub[(long long)rid * 64 + d] = ks[src];
  vsub[(long long)rid * 64 + d] = vs[src];
}

// ---------------- LSE merges ----------------
__device__ __forceinline__ float lse_merge(float l1, float l2) {
  return fmaxf(l1, l2) + log1pf(expf(-fabsf(l1 - l2)));
}

// merge block-diag + residual in sorted order, then unsort -> un0 buffer
__global__ void combine_unsort_kernel(const float* __restrict__ abA, const float* __restrict__ abL,
                                      const float* __restrict__ arA, const float* __restrict__ arL,
                                      const int* __restrict__ sinv_q,
                                      float* __restrict__ oA, float* __restrict__ oL) {
  int rid = blockIdx.x;           // HEADS*NSORT
  int d = threadIdx.x;
  int h = rid / NSORT, i = rid % NSORT;
  int p = sinv_q[(long long)h * NSORT + i];
  long long src = (long long)h * NPAD + p;
  float l1 = abL[src], l2 = arL[src];
  float c = 1.f / (1.f + expf(l2 - l1));
  float a = c * abA[src * 64 + d] + (1.f - c) * arA[src * 64 + d];
  long long dst = (long long)h * NSORT + i;
  oA[dst * 64 + d] = a;
  if (d == 0) oL[dst] = lse_merge(l1, l2);
}

// in-place merge: tgt = combine(tgt, src), with unit-stride row mapping
__global__ void combine_inplace_kernel(float* __restrict__ tA, float* __restrict__ tL,
                                       long long t_hs, long long t_off, long long t_ss,
                                       const float* __restrict__ sA, const float* __restrict__ sL,
                                       long long s_hs, long long s_off, long long s_ss,
                                       int inner, int rows) {
  int rid = blockIdx.x;
  int d = threadIdx.x;
  int r = rid % rows;
  int t = rid / rows;
  int s = t % inner;
  int h = t / inner;
  long long ti = (long long)h * t_hs + t_off + (long long)s * t_ss + r;
  long long si = (long long)h * s_hs + s_off + (long long)s * s_ss + r;
  float l1 = tL[ti], l2 = sL[si];
  float c = 1.f / (1.f + expf(l2 - l1));
  float a = c * tA[ti * 64 + d] + (1.f - c) * sA[si * 64 + d];
  tA[ti * 64 + d] = a;
  if (d == 0) tL[ti] = lse_merge(l1, l2);
}

// final: combine prefix with tail (tail lse = -inf for first SINK rows)
__global__ void final_kernel(const float* __restrict__ pA, const float* __restrict__ pL,
                             const float* __restrict__ tA, const float* __restrict__ tL,
                             float* __restrict__ out) {
  int rid = blockIdx.x;           // HEADS*SEQ
  int d = threadIdx.x;
  int h = rid >> 12, n = rid & 4095;
  long long pi = (long long)h * SEQ + n;
  float v;
  if (n < SINK) {
    v = pA[pi * 64 + d];
  } else {
    long long ti = (long long)h * TAIL + (n - SINK);
    float l1 = pL[pi], l2 = tL[ti];
    float c = 1.f / (1.f + expf(l2 - l1));
    v = c * pA[pi * 64 + d] + (1.f - c) * tA[ti * 64 + d];
  }
  out[pi * 64 + d] = v;
}

// ---------------- host launcher ----------------
extern "C" void kernel_launch(void* const* d_in, const int* in_sizes, int n_in,
                              void* d_out, int out_size, void* d_ws, size_t ws_size,
                              hipStream_t stream) {
  const float* Q    = (const float*)d_in[0];
  const float* K    = (const float*)d_in[1];
  const float* V    = (const float*)d_in[2];
  const float* proj = (const float*)d_in[3];
  float* out = (float*)d_out;
  (void)in_sizes; (void)n_in; (void)out_size; (void)ws_size;

  size_t off = 0;
  auto alloc = [&](size_t bytes) -> void* {
    void* p = (char*)d_ws + off;
    off += (bytes + 255) & ~(size_t)255;
    return p;
  };
  float* prefixA = (float*)alloc((size_t)HEADS * SEQ * 64 * 4);
  float* prefixL = (float*)alloc((size_t)HEADS * SEQ * 4);
  float* tailA   = (float*)alloc((size_t)HEADS * TAIL * 64 * 4);
  float* tailL   = (float*)alloc((size_t)HEADS * TAIL * 4);
  float* un1A    = (float*)alloc((size_t)HEADS * HALF0 * 64 * 4);
  float* un1L    = (float*)alloc((size_t)HEADS * HALF0 * 4);
  float* un0A    = (float*)alloc((size_t)HEADS * NSORT * 64 * 4);
  float* un0L    = (float*)alloc((size_t)HEADS * NSORT * 4);
  float* qs      = (float*)alloc((size_t)HEADS * NPAD * 64 * 4);
  float* ks      = (float*)alloc((size_t)HEADS * NPAD * 64 * 4);
  float* vs      = (float*)alloc((size_t)HEADS * NPAD * 64 * 4);
  float* abA     = (float*)alloc((size_t)HEADS * NPAD * 64 * 4);
  float* abL     = (float*)alloc((size_t)HEADS * NPAD * 4);
  float* arA     = (float*)alloc((size_t)HEADS * NPAD * 64 * 4);
  float* arL     = (float*)alloc((size_t)HEADS * NPAD * 4);
  float* ksub    = (float*)alloc((size_t)HEADS * SAMP * 64 * 4);
  float* vsub    = (float*)alloc((size_t)HEADS * SAMP * 64 * 4);
  int* codes_q   = (int*)alloc((size_t)HEADS * NSORT * 4);
  int* codes_k   = (int*)alloc((size_t)HEADS * NSORT * 4);
  int* sidx_q    = (int*)alloc((size_t)HEADS * NPAD * 4);
  int* sidx_k    = (int*)alloc((size_t)HEADS * NPAD * 4);
  int* sinv_q    = (int*)alloc((size_t)HEADS * NSORT * 4);
  int* sinv_k    = (int*)alloc((size_t)HEADS * NSORT * 4);
  int* sampled   = (int*)alloc((size_t)HEADS * SAMP * 4);
  int* kblk      = (int*)alloc((size_t)HEADS * SAMP * 4);

  const float scale = 0.125f;   // 1/sqrt(64)

  // ---- LSH pipeline (level-0 halves: q = tail[2032:4064) -> global off 2064; k = tail[0:2032) -> off 32)
  lsh_hash_kernel<<<dim3(8, HEADS), 256, 0, stream>>>(Q, proj, codes_q, NSORT, SINK + HALF0);
  lsh_hash_kernel<<<dim3(8, HEADS), 256, 0, stream>>>(K, proj, codes_k, NSORT, SINK);
  argsort_kernel<<<HEADS, 256, 0, stream>>>(codes_q, sidx_q, sinv_q, NSORT, NPAD);
  argsort_kernel<<<HEADS, 256, 0, stream>>>(codes_k, sidx_k, sinv_k, NSORT, NPAD);
  gather_rows_kernel<<<HEADS * NPAD, 64, 0, stream>>>(Q, sidx_q, qs, NSORT, NPAD, SINK + HALF0);
  gather_rows_kernel<<<HEADS * NPAD, 64, 0, stream>>>(K, sidx_k, ks, NSORT, NPAD, SINK);
  gather_rows_kernel<<<HEADS * NPAD, 64, 0, stream>>>(V, sidx_k, vs, NSORT, NPAD, SINK);
  sample_kernel<<<(HEADS * SAMP + 255) / 256, 256, 0, stream>>>(sampled, kblk);
  gather_sub_kernel<<<HEADS * SAMP, 64, 0, stream>>>(ks, vs, sampled, ksub, vsub);

  // ---- prefix: causal, all 4096 q vs first 32 k
  flash_attn_kernel<<<dim3(64, HEADS), 128, 0, stream>>>(
      Q, K, V, prefixA, prefixL, SEQ, SINK, 1,
      SEQ, 0, 0,  SEQ, 0, 0,  SEQ, 0, 0,  SEQ, 0, 0,
      1, 0, nullptr, scale, 0.f);

  // ---- level-2 block-diagonal causal: 64 units of 1016
  flash_attn_kernel<<<dim3(16, HEADS * 4), 128, 0, stream>>>(
      Q, K, V, tailA, tailL, HALF1, HALF1, 4,
      SEQ, SINK, HALF1,  SEQ, SINK, HALF1,
      TAIL, 0, HALF1,    TAIL, 0, HALF1,
      1, 0, nullptr, scale, 0.f);

  // ---- level-1 un-masked: q=[1016:2032) vs k=[0:1016) of each 2032-segment
  flash_attn_kernel<<<dim3(16, HEADS * 2), 128, 0, stream>>>(
      Q, K, V, un1A, un1L, HALF1, HALF1, 2,
      SEQ, SINK + HALF1, HALF0,  SEQ, SINK, HALF0,
      HALF0, 0, HALF1,           HALF0, 0, HALF1,
      0, 0, nullptr, scale, 0.f);

  // ---- LSH block-diagonal attention: 8 blocks of 256x256 per head (sorted, padded)
  flash_attn_kernel<<<dim3(4, HEADS * NBLK), 128, 0, stream>>>(
      qs, ks, vs, abA, abL, 256, 256, NBLK,
      NPAD, 0, 256,  NPAD, 0, 256,
      NPAD, 0, 256,  NPAD, 0, 256,
      0, 0, nullptr, scale, 0.f);

  // ---- sampled residual: 2032 sorted q vs 256 sampled keys, same-block bias masked
  flash_attn_kernel<<<dim3(32, HEADS), 128, 0, stream>>>(
      qs, ksub, vsub, arA, arL, NSORT, SAMP, 1,
      NPAD, 0, 0,  SAMP, 0, 0,
      NPAD, 0, 0,  NPAD, 0, 0,
      0, 1, kblk, scale, logf((float)NSORT / (float)SAMP));

  // ---- merge block + residual, unsort -> un0
  combine_unsort_kernel<<<HEADS * NSORT, 64, 0, stream>>>(abA, abL, arA, arL, sinv_q, un0A, un0L);

  // ---- merge un1 into tail lower halves of each level-1 segment
  combine_inplace_kernel<<<HEADS * HALF0, 64, 0, stream>>>(
      tailA, tailL, TAIL, HALF1, HALF0,
      un1A, un1L,  HALF0, 0, HALF1,
      2, HALF1);

  // ---- merge un0 into tail rows [2032, 4064)
  combine_inplace_kernel<<<HEADS * NSORT, 64, 0, stream>>>(
      tailA, tailL, TAIL, HALF0, 0,
      un0A, un0L,  NSORT, 0, 0,
      1, NSORT);

  // ---- final merge with prefix
  final_kernel<<<HEADS * SEQ, 64, 0, stream>>>(prefixA, prefixL, tailA, tailL, out);
}
